// InteractionModelingModule_14766097563802
// MI455X (gfx1250) — matrix operand order repaired and checked by reading, not compiled, not run on hardware
//
#include <hip/hip_runtime.h>
#include <hip/hip_bf16.h>

#define B_ 128
#define N_ 128
#define P_ 1024
#define H_ 768
#define KB_ (H_ / 32)          // 24 k-blocks
#define LDA_ 40                // halves per LDS row (padded)
#define NEGV (-1e9f)

typedef __attribute__((ext_vector_type(16))) __bf16 v16bf;
typedef __attribute__((ext_vector_type(8)))  float  v8f;

union FragBF { v16bf v; uint4 q[2]; };

__device__ inline uint2 cvt4_bf16(float4 f) {
    union { __bf16 h[4]; uint2 u; } r;
    r.h[0] = (__bf16)f.x; r.h[1] = (__bf16)f.y;
    r.h[2] = (__bf16)f.z; r.h[3] = (__bf16)f.w;
    return r.u;
}

__device__ inline float wave_max(float v) {
    #pragma unroll
    for (int o = 16; o > 0; o >>= 1) v = fmaxf(v, __shfl_down(v, o, 32));
    return v;
}
__device__ inline float wave_sum(float v) {
    #pragma unroll
    for (int o = 16; o > 0; o >>= 1) v += __shfl_down(v, o, 32);
    return v;
}

// ---------------------------------------------------------------------------
// Stage 1: X = H_title @ W_c^T  (M = B*N = 16384, K = H, Ncol = H), bf16 out.
// 128x128 tile per 256-thread block, LDS double-buffered, 8 WMMA frags/wave.
// ---------------------------------------------------------------------------
__global__ __launch_bounds__(256)
void k_gemm_xw(const float* __restrict__ Ht, const float* __restrict__ Wc,
               __bf16* __restrict__ X) {
    const int rowTile = blockIdx.x;   // 0..127
    const int colTile = blockIdx.y;   // 0..5
    const int tid  = threadIdx.x;
    const int lane = tid & 31, wid = tid >> 5;
    const int wrow = wid >> 1;        // 32-row band
    const int wcol = wid & 1;         // 64-col band

    __shared__ __align__(16) __bf16 As[2][128 * LDA_];
    __shared__ __align__(16) __bf16 Bs[2][128 * LDA_];

    v8f acc[2][4];
    #pragma unroll
    for (int i = 0; i < 2; i++)
        #pragma unroll
        for (int j = 0; j < 4; j++)
            #pragma unroll
            for (int e = 0; e < 8; e++) acc[i][j][e] = 0.f;

    const int m = lane & 15, g = lane >> 4;
    float4 ra[4], rb[4];

    // ---- preload k-block 0 into registers, stage into buffer 0
    #pragma unroll
    for (int i = 0; i < 4; i++) {
        int idx = tid + i * 256, r = idx >> 3, c = idx & 7;
        ra[i] = *reinterpret_cast<const float4*>(
            Ht + (size_t)(rowTile * 128 + r) * H_ + c * 4);
        rb[i] = *reinterpret_cast<const float4*>(
            Wc + (size_t)(colTile * 128 + r) * H_ + c * 4);
    }
    #pragma unroll
    for (int i = 0; i < 4; i++) {
        int idx = tid + i * 256, r = idx >> 3, c = idx & 7;
        *reinterpret_cast<uint2*>(&As[0][r * LDA_ + c * 4]) = cvt4_bf16(ra[i]);
        *reinterpret_cast<uint2*>(&Bs[0][r * LDA_ + c * 4]) = cvt4_bf16(rb[i]);
    }
    __syncthreads();

    int buf = 0;
    for (int kb = 0; kb < KB_; kb++) {
        const bool more = (kb + 1) < KB_;
        const int k0n = (kb + 1) * 32;
        if (more) {   // issue next tile's global loads before the WMMA chain
            #pragma unroll
            for (int i = 0; i < 4; i++) {
                int idx = tid + i * 256, r = idx >> 3, c = idx & 7;
                ra[i] = *reinterpret_cast<const float4*>(
                    Ht + (size_t)(rowTile * 128 + r) * H_ + k0n + c * 4);
                rb[i] = *reinterpret_cast<const float4*>(
                    Wc + (size_t)(colTile * 128 + r) * H_ + k0n + c * 4);
            }
        }

        const __bf16* Ab = As[buf];
        const __bf16* Bb = Bs[buf];
        FragBF a[2], b[4];
        #pragma unroll
        for (int mi = 0; mi < 2; mi++) {
            const __bf16* p = &Ab[(wrow * 32 + mi * 16 + m) * LDA_ + 8 * g];
            a[mi].q[0] = *reinterpret_cast<const uint4*>(p);       // k = 8g..
            a[mi].q[1] = *reinterpret_cast<const uint4*>(p + 16);  // k = 16+8g..
        }
        #pragma unroll
        for (int ni = 0; ni < 4; ni++) {
            const __bf16* p = &Bb[(wcol * 64 + ni * 16 + m) * LDA_ + 8 * g];
            b[ni].q[0] = *reinterpret_cast<const uint4*>(p);       // k = 8g..
            b[ni].q[1] = *reinterpret_cast<const uint4*>(p + 16);
        }
        #pragma unroll
        for (int mi = 0; mi < 2; mi++)
            #pragma unroll
            for (int ni = 0; ni < 4; ni++)
                acc[mi][ni] = __builtin_amdgcn_wmma_f32_16x16x32_bf16(
                    false, a[mi].v, false, b[ni].v, (short)0, acc[mi][ni],
                    false, false);

        if (more) {   // stage next tile into the other buffer
            #pragma unroll
            for (int i = 0; i < 4; i++) {
                int idx = tid + i * 256, r = idx >> 3, c = idx & 7;
                *reinterpret_cast<uint2*>(&As[buf ^ 1][r * LDA_ + c * 4]) = cvt4_bf16(ra[i]);
                *reinterpret_cast<uint2*>(&Bs[buf ^ 1][r * LDA_ + c * 4]) = cvt4_bf16(rb[i]);
            }
        }
        __syncthreads();
        buf ^= 1;
    }

    #pragma unroll
    for (int mi = 0; mi < 2; mi++)
        #pragma unroll
        for (int ni = 0; ni < 4; ni++) {
            int col = colTile * 128 + wcol * 64 + ni * 16 + m;
            #pragma unroll
            for (int r = 0; r < 8; r++) {
                int row = rowTile * 128 + wrow * 32 + mi * 16 + r + 8 * g;
                X[(size_t)row * H_ + col] = (__bf16)acc[mi][ni][r];
            }
        }
}

// ---------------------------------------------------------------------------
// Stage 2: S[b] = X[b] (128xH bf16) @ H_lead[b]^T (1024xH f32->bf16), f32 out.
// A-tile is a raw bf16 copy -> global_load_async_to_lds_b128 (ASYNCcnt path).
// B-tile needs f32->bf16 conversion -> register staging. Double-buffered LDS.
// ---------------------------------------------------------------------------
__global__ __launch_bounds__(256)
void k_gemm_s(const __bf16* __restrict__ X, const float* __restrict__ Hl,
              float* __restrict__ S) {
    const int pTile = blockIdx.x;   // 0..7
    const int b     = blockIdx.y;   // 0..127
    const int tid  = threadIdx.x;
    const int lane = tid & 31, wid = tid >> 5;
    const int wrow = wid >> 1, wcol = wid & 1;

    __shared__ __align__(16) __bf16 As[2][128 * LDA_];
    __shared__ __align__(16) __bf16 Bs[2][128 * LDA_];

    v8f acc[2][4];
    #pragma unroll
    for (int i = 0; i < 2; i++)
        #pragma unroll
        for (int j = 0; j < 4; j++)
            #pragma unroll
            for (int e = 0; e < 8; e++) acc[i][j][e] = 0.f;

    const int m = lane & 15, g = lane >> 4;
    float4 rb[4];

    // per-thread A-copy chunk coordinates (2 x 16B per thread)
    const int ar0 = (tid + 0)   >> 2, ac0 = (tid + 0)   & 3;
    const int ar1 = (tid + 256) >> 2, ac1 = (tid + 256) & 3;

    // generic->LDS: low 32 bits of the flat address are the wave-relative LDS offset
    const unsigned ldsA0 = (unsigned)(size_t)&As[0][0];
    const unsigned ldsA1 = (unsigned)(size_t)&As[1][0];
    const unsigned aoff0 = (unsigned)((ar0 * LDA_ + ac0 * 8) * sizeof(__bf16));
    const unsigned aoff1 = (unsigned)((ar1 * LDA_ + ac1 * 8) * sizeof(__bf16));
    const unsigned gA0   = (unsigned)(((size_t)(b * N_ + ar0) * H_ + ac0 * 8) * sizeof(__bf16));
    const unsigned gA1   = (unsigned)(((size_t)(b * N_ + ar1) * H_ + ac1 * 8) * sizeof(__bf16));

    // ---- k-block 0: async A -> As[0], staged B -> Bs[0]
    asm volatile("global_load_async_to_lds_b128 %0, %1, %2"
                 :: "v"(ldsA0 + aoff0), "v"(gA0), "s"(X) : "memory");
    asm volatile("global_load_async_to_lds_b128 %0, %1, %2"
                 :: "v"(ldsA0 + aoff1), "v"(gA1), "s"(X) : "memory");
    #pragma unroll
    for (int i = 0; i < 4; i++) {
        int idx = tid + i * 256, r = idx >> 3, c = idx & 7;
        rb[i] = *reinterpret_cast<const float4*>(
            Hl + (size_t)(b * P_ + pTile * 128 + r) * H_ + c * 4);
    }
    #pragma unroll
    for (int i = 0; i < 4; i++) {
        int idx = tid + i * 256, r = idx >> 3, c = idx & 7;
        *reinterpret_cast<uint2*>(&Bs[0][r * LDA_ + c * 4]) = cvt4_bf16(rb[i]);
    }
    asm volatile("s_wait_asynccnt 0x0" ::: "memory");
    __syncthreads();

    int buf = 0;
    for (int kb = 0; kb < KB_; kb++) {
        const bool more = (kb + 1) < KB_;
        const int k0n = (kb + 1) * 32;
        if (more) {
            const unsigned ldsAn = (buf ? ldsA0 : ldsA1);
            const unsigned gk = (unsigned)(k0n * sizeof(__bf16));
            asm volatile("global_load_async_to_lds_b128 %0, %1, %2"
                         :: "v"(ldsAn + aoff0), "v"(gA0 + gk), "s"(X) : "memory");
            asm volatile("global_load_async_to_lds_b128 %0, %1, %2"
                         :: "v"(ldsAn + aoff1), "v"(gA1 + gk), "s"(X) : "memory");
            #pragma unroll
            for (int i = 0; i < 4; i++) {
                int idx = tid + i * 256, r = idx >> 3, c = idx & 7;
                rb[i] = *reinterpret_cast<const float4*>(
                    Hl + (size_t)(b * P_ + pTile * 128 + r) * H_ + k0n + c * 4);
            }
        }

        const __bf16* Ab = As[buf];
        const __bf16* Bb = Bs[buf];
        FragBF a[2], bq[4];
        #pragma unroll
        for (int mi = 0; mi < 2; mi++) {
            const __bf16* p = &Ab[(wrow * 32 + mi * 16 + m) * LDA_ + 8 * g];
            a[mi].q[0] = *reinterpret_cast<const uint4*>(p);
            a[mi].q[1] = *reinterpret_cast<const uint4*>(p + 16);
        }
        #pragma unroll
        for (int ni = 0; ni < 4; ni++) {
            const __bf16* p = &Bb[(wcol * 64 + ni * 16 + m) * LDA_ + 8 * g];
            bq[ni].q[0] = *reinterpret_cast<const uint4*>(p);
            bq[ni].q[1] = *reinterpret_cast<const uint4*>(p + 16);
        }
        #pragma unroll
        for (int mi = 0; mi < 2; mi++)
            #pragma unroll
            for (int ni = 0; ni < 4; ni++)
                acc[mi][ni] = __builtin_amdgcn_wmma_f32_16x16x32_bf16(
                    false, a[mi].v, false, bq[ni].v, (short)0, acc[mi][ni],
                    false, false);

        if (more) {
            #pragma unroll
            for (int i = 0; i < 4; i++) {
                int idx = tid + i * 256, r = idx >> 3, c = idx & 7;
                *reinterpret_cast<uint2*>(&Bs[buf ^ 1][r * LDA_ + c * 4]) = cvt4_bf16(rb[i]);
            }
            asm volatile("s_wait_asynccnt 0x0" ::: "memory");
        }
        __syncthreads();
        buf ^= 1;
    }

    #pragma unroll
    for (int mi = 0; mi < 2; mi++)
        #pragma unroll
        for (int ni = 0; ni < 4; ni++) {
            int col = pTile * 128 + wcol * 64 + ni * 16 + m;
            #pragma unroll
            for (int r = 0; r < 8; r++) {
                int row = wrow * 32 + mi * 16 + r + 8 * g;
                S[((size_t)b * N_ + row) * P_ + col] = acc[mi][ni][r];
            }
        }
}

// ---------------------------------------------------------------------------
// Stage 3a: row softmax over P (lead mask), reduced to v_T[p] = sum_n tm[n]*A_T
// ---------------------------------------------------------------------------
__global__ __launch_bounds__(256)
void k_soft_rows(const float* __restrict__ S, const int* __restrict__ tmask,
                 const int* __restrict__ lmask, float* __restrict__ vT) {
    const int b = blockIdx.x, t = threadIdx.x;
    const int lane = t & 31, wid = t >> 5;
    __shared__ float lmf[P_];
    __shared__ float vacc[P_];
    __shared__ float tmf[N_];
    __shared__ float redm[8], reds[8];

    #pragma unroll
    for (int i = 0; i < 4; i++) {
        int p = t + i * 256;
        lmf[p]  = (float)lmask[(size_t)b * P_ + p];
        vacc[p] = 0.f;
    }
    if (t < N_) tmf[t] = (float)tmask[(size_t)b * N_ + t];
    __syncthreads();

    for (int n = 0; n < N_; n++) {
        float s[4]; float mx = NEGV;
        #pragma unroll
        for (int i = 0; i < 4; i++) {
            int p = t + i * 256;
            float sv = S[((size_t)b * N_ + n) * P_ + p];
            s[i] = (lmf[p] > 0.5f) ? sv : NEGV;
            mx = fmaxf(mx, s[i]);
        }
        mx = wave_max(mx);
        if (lane == 0) redm[wid] = mx;
        __syncthreads();
        float mrow = NEGV;
        #pragma unroll
        for (int i = 0; i < 8; i++) mrow = fmaxf(mrow, redm[i]);

        float e[4]; float ls = 0.f;
        #pragma unroll
        for (int i = 0; i < 4; i++) { e[i] = __expf(s[i] - mrow); ls += e[i]; }
        ls = wave_sum(ls);
        if (lane == 0) reds[wid] = ls;
        __syncthreads();
        float Z = 0.f;
        #pragma unroll
        for (int i = 0; i < 8; i++) Z += reds[i];

        float w = tmf[n] / Z;
        #pragma unroll
        for (int i = 0; i < 4; i++) vacc[t + i * 256] += w * e[i];
        __syncthreads();
    }
    #pragma unroll
    for (int i = 0; i < 4; i++) {
        int p = t + i * 256;
        vT[(size_t)b * P_ + p] = vacc[p];
    }
}

// ---------------------------------------------------------------------------
// Stage 3b: col softmax over N (title mask), reduced to w_L[n] = sum_p lm[p]*A_L
// ---------------------------------------------------------------------------
__global__ __launch_bounds__(256)
void k_soft_cols(const float* __restrict__ S, const int* __restrict__ tmask,
                 const int* __restrict__ lmask, float* __restrict__ wL) {
    const int b = blockIdx.x, t = threadIdx.x;
    const int lane = t & 31;
    __shared__ float tmf[N_];
    __shared__ float wacc[N_];
    if (t < N_) { tmf[t] = (float)tmask[(size_t)b * N_ + t]; wacc[t] = 0.f; }
    __syncthreads();

    float mu[4], iw[4];
    #pragma unroll
    for (int i = 0; i < 4; i++) {
        int p = t + i * 256;
        float mx = NEGV;
        for (int n = 0; n < N_; n++) {
            float sv = S[((size_t)b * N_ + n) * P_ + p];
            mx = fmaxf(mx, (tmf[n] > 0.5f) ? sv : NEGV);
        }
        mu[i] = mx;
    }
    #pragma unroll
    for (int i = 0; i < 4; i++) {
        int p = t + i * 256;
        float W = 0.f;
        for (int n = 0; n < N_; n++) {
            float sv = S[((size_t)b * N_ + n) * P_ + p];
            W += __expf(((tmf[n] > 0.5f) ? sv : NEGV) - mu[i]);
        }
        iw[i] = (float)lmask[(size_t)b * P_ + p] / W;
    }
    for (int n = 0; n < N_; n++) {
        float c = 0.f;
        #pragma unroll
        for (int i = 0; i < 4; i++) {
            int p = t + i * 256;
            float sv = S[((size_t)b * N_ + n) * P_ + p];
            c += iw[i] * __expf(((tmf[n] > 0.5f) ? sv : NEGV) - mu[i]);
        }
        c = wave_sum(c);
        if (lane == 0) atomicAdd(&wacc[n], c);
    }
    __syncthreads();
    if (t < N_) wL[(size_t)b * N_ + t] = wacc[t];
}

// ---------------------------------------------------------------------------
// Stage 4: pooled outputs. one block per batch; thread owns 3 feature columns
// ---------------------------------------------------------------------------
__global__ __launch_bounds__(256)
void k_pool(const float* __restrict__ Ht, const float* __restrict__ Hl,
            const int* __restrict__ tmask, const int* __restrict__ lmask,
            const float* __restrict__ vT, const float* __restrict__ wL,
            float* __restrict__ out) {
    const int b = blockIdx.x, t = threadIdx.x;
    __shared__ float tmf[N_], wl[N_];
    __shared__ float lmf[P_], vt[P_];
    if (t < N_) { tmf[t] = (float)tmask[(size_t)b * N_ + t]; wl[t] = wL[(size_t)b * N_ + t]; }
    #pragma unroll
    for (int i = 0; i < 4; i++) {
        int p = t + i * 256;
        lmf[p] = (float)lmask[(size_t)b * P_ + p];
        vt[p]  = vT[(size_t)b * P_ + p];
    }
    __syncthreads();

    float Lt = 0.f; for (int n = 0; n < N_; n++) Lt += tmf[n];
    float Ll = 0.f; for (int p = 0; p < P_; p++) Ll += lmf[p];
    Lt = fmaxf(Lt, 1e-9f); Ll = fmaxf(Ll, 1e-9f);

    float at[3] = {0,0,0}, acl[3] = {0,0,0}, al[3] = {0,0,0}, act[3] = {0,0,0};
    for (int n = 0; n < N_; n++) {
        float tw = tmf[n], ww = wl[n];
        const float* r = Ht + ((size_t)b * N_ + n) * H_;
        #pragma unroll
        for (int j = 0; j < 3; j++) { float h = r[t + j * 256]; at[j] += tw * h; acl[j] += ww * h; }
    }
    for (int p = 0; p < P_; p++) {
        float lw = lmf[p], vw = vt[p];
        const float* r = Hl + ((size_t)b * P_ + p) * H_;
        #pragma unroll
        for (int j = 0; j < 3; j++) { float h = r[t + j * 256]; al[j] += lw * h; act[j] += vw * h; }
    }
    const size_t OUT2 = (size_t)B_ * 2 * H_;
    #pragma unroll
    for (int j = 0; j < 3; j++) {
        int d = t + j * 256;
        out[(size_t)b * 2 * H_ + d]             = at[j] / Lt;   // r_title[:, :H]
        out[(size_t)b * 2 * H_ + H_ + d]        = act[j] / Lt;  // r_title[:, H:]
        out[OUT2 + (size_t)b * 2 * H_ + d]      = al[j] / Ll;   // r_lead[:, :H]
        out[OUT2 + (size_t)b * 2 * H_ + H_ + d] = acl[j] / Ll;  // r_lead[:, H:]
    }
}

extern "C" void kernel_launch(void* const* d_in, const int* in_sizes, int n_in,
                              void* d_out, int out_size, void* d_ws, size_t ws_size,
                              hipStream_t stream) {
    const float* Ht    = (const float*)d_in[0];
    const int*   tmask = (const int*)d_in[1];
    const float* Hl    = (const float*)d_in[2];
    const int*   lmask = (const int*)d_in[3];
    const float* Wc    = (const float*)d_in[4];
    float* out = (float*)d_out;

    char* ws = (char*)d_ws;
    const size_t X_BYTES = (size_t)B_ * N_ * H_ * 2;      // 25,165,824
    const size_t S_BYTES = (size_t)B_ * N_ * P_ * 4;      // 67,108,864
    const size_t V_BYTES = (size_t)B_ * P_ * 4;           //    524,288
    __bf16* X = (__bf16*)ws;
    float*  S = (float*)(ws + X_BYTES);
    float* vT = (float*)(ws + X_BYTES + S_BYTES);
    float* wL = (float*)(ws + X_BYTES + S_BYTES + V_BYTES);

    dim3 g1((B_ * N_) / 128, H_ / 128);   // 128 x 6
    k_gemm_xw<<<g1, 256, 0, stream>>>(Ht, Wc, X);

    dim3 g2(P_ / 128, B_);                // 8 x 128
    k_gemm_s<<<g2, 256, 0, stream>>>(X, Hl, S);

    k_soft_rows<<<B_, 256, 0, stream>>>(S, tmask, lmask, vT);
    k_soft_cols<<<B_, 256, 0, stream>>>(S, tmask, lmask, wL);
    k_pool<<<B_, 256, 0, stream>>>(Ht, Hl, tmask, lmask, vT, wL, out);
}
